// MoELayer_56186762166280
// MI455X (gfx1250) — matrix operand order repaired and checked by reading, not compiled, not run on hardware
//
#include <hip/hip_runtime.h>
#include <cstdint>

#define B_ 4
#define S_ 1024
#define D_ 1024
#define I_ 4096
#define E_ 8
#define T_ (B_*S_)      // 4096 tokens
#define IC 128          // I-chunk per FFN main-loop iteration

typedef __bf16 bf16;
typedef __attribute__((ext_vector_type(16))) __bf16 v16bf;
typedef __attribute__((ext_vector_type(8)))  float  v8f;

// ---------------------------------------------------------------- utilities

// A-operand fragment (16x32 MxK bf16), contiguous k order, same as the B repack:
// lanes 0-15: M=lane, elems [K0..K15]; lanes 16-31: M=lane-16, elems [K16..K31]
__device__ __forceinline__ v16bf load_a_frag(const bf16* base, int rowStride, int m, int k0, bool hi) {
    union { v16bf v; uint4 q[2]; } u;
    const bf16* p = base + m * rowStride + k0 + (hi ? 16 : 0);
    u.q[0] = *(const uint4*)(p);        // 8 bf16 = 16B
    u.q[1] = *(const uint4*)(p + 8);    // next 8 bf16, K+8
    return u.v;
}

// B-operand fragment from the prepacked layout: 32 contiguous bytes per lane.
__device__ __forceinline__ v16bf load_b_frag(const bf16* fragBase, int lane) {
    union { v16bf v; uint4 q[2]; } u;
    const uint4* p = (const uint4*)(fragBase + lane * 16);
    u.q[0] = p[0];
    u.q[1] = p[1];
    return u.v;
}

__device__ __forceinline__ v8f v8f_zero() {
    v8f z = {0.f,0.f,0.f,0.f,0.f,0.f,0.f,0.f};
    return z;
}

// ---------------------------------------------------------------- kernel 1: zero out + counters

__global__ void k_zero(float* __restrict__ dout, int* __restrict__ counts) {
    size_t n = (size_t)T_ * D_ + 1;
    size_t stride = (size_t)gridDim.x * blockDim.x;
    for (size_t i = (size_t)blockIdx.x * blockDim.x + threadIdx.x; i < n; i += stride)
        dout[i] = 0.0f;
    if (blockIdx.x == 0 && threadIdx.x < E_) counts[threadIdx.x] = 0;
}

// ---------------------------------------------------------------- kernels 2/3: weight repack f32 -> bf16 fragments

// w1 layout (E, D, I): K = d index, N = i index.
// dst: [e][i_tile(I/16)][kstep(D/32)][lane(32)][16] (512 bf16 = 1KB per fragment)
__global__ void k_repack_w1(const float* __restrict__ w1, bf16* __restrict__ w1bf) {
    size_t N = (size_t)E_ * D_ * I_;
    size_t stride = (size_t)gridDim.x * blockDim.x;
    for (size_t idx = (size_t)blockIdx.x * blockDim.x + threadIdx.x; idx < N; idx += stride) {
        int e = (int)(idx / ((size_t)D_ * I_));
        size_t rem = idx % ((size_t)D_ * I_);
        int k = (int)(rem / I_);
        int i = (int)(rem % I_);
        int iTile = i >> 4, n = i & 15;
        int ks = k >> 5, kr = k & 31;
        int lane = (kr < 16) ? n : (n + 16);
        int j = kr & 15;
        size_t off = ((((size_t)e * (I_/16) + iTile) * (D_/32) + ks) * 32 + lane) * 16 + j;
        w1bf[off] = (bf16)w1[idx];
    }
}

// w2 layout (E, I, D): K = i index, N = d index.
// dst: [e][d_tile(D/16)][kstep(I/32)][lane(32)][16]
__global__ void k_repack_w2(const float* __restrict__ w2, bf16* __restrict__ w2bf) {
    size_t N = (size_t)E_ * I_ * D_;
    size_t stride = (size_t)gridDim.x * blockDim.x;
    for (size_t idx = (size_t)blockIdx.x * blockDim.x + threadIdx.x; idx < N; idx += stride) {
        int e = (int)(idx / ((size_t)I_ * D_));
        size_t rem = idx % ((size_t)I_ * D_);
        int i = (int)(rem / D_);
        int d = (int)(rem % D_);
        int dTile = d >> 4, n = d & 15;
        int ks = i >> 5, kr = i & 31;
        int lane = (kr < 16) ? n : (n + 16);
        int j = kr & 15;
        size_t off = ((((size_t)e * (D_/16) + dTile) * (I_/32) + ks) * 32 + lane) * 16 + j;
        w2bf[off] = (bf16)w2[idx];
    }
}

// ---------------------------------------------------------------- kernel 4: gating (one wave per token)

__global__ __launch_bounds__(256) void k_gate(const float* __restrict__ h,
                                              const float* __restrict__ gate_w,
                                              bf16* __restrict__ hbf,
                                              int* __restrict__ counts,
                                              int* __restrict__ tok_list,
                                              float* __restrict__ wt_list) {
    int wave = threadIdx.x >> 5;
    int lane = threadIdx.x & 31;
    int t = blockIdx.x * 8 + wave;
    if (t >= T_) return;

    float p[E_];
#pragma unroll
    for (int e = 0; e < E_; ++e) p[e] = 0.f;

    const float* hr = h + (size_t)t * D_;
    for (int d = lane; d < D_; d += 32) {
        float hv = hr[d];
        hbf[(size_t)t * D_ + d] = (bf16)hv;
#pragma unroll
        for (int e = 0; e < E_; ++e) p[e] += hv * gate_w[e * D_ + d];
    }
#pragma unroll
    for (int off = 16; off > 0; off >>= 1) {
#pragma unroll
        for (int e = 0; e < E_; ++e) p[e] += __shfl_xor(p[e], off, 32);
    }

    if (lane == 0) {
        float mx = p[0];
#pragma unroll
        for (int e = 1; e < E_; ++e) mx = fmaxf(mx, p[e]);
        float pr[E_];
#pragma unroll
        for (int e = 0; e < E_; ++e) pr[e] = __expf(p[e] - mx);
        int s0 = 0;
#pragma unroll
        for (int e = 1; e < E_; ++e) if (pr[e] > pr[s0]) s0 = e;
        int s1 = (s0 == 0) ? 1 : 0;
#pragma unroll
        for (int e = 0; e < E_; ++e) if (e != s0 && pr[e] > pr[s1]) s1 = e;
        float sum2 = pr[s0] + pr[s1];
        float w0 = pr[s0] / sum2;
        float w1v = pr[s1] / sum2;

        int slot0 = atomicAdd(&counts[s0], 1);
        tok_list[s0 * T_ + slot0] = t;
        wt_list[s0 * T_ + slot0] = w0;
        int slot1 = atomicAdd(&counts[s1], 1);
        tok_list[s1 * T_ + slot1] = t;
        wt_list[s1 * T_ + slot1] = w1v;
    }
}

// ---------------------------------------------------------------- kernel 5: expert FFN (WMMA)

__global__ __launch_bounds__(256) void k_ffn(const bf16* __restrict__ hbf,
                                             const bf16* __restrict__ w1bf,
                                             const bf16* __restrict__ w2bf,
                                             const float* __restrict__ b1,
                                             const float* __restrict__ b2,
                                             const int* __restrict__ counts,
                                             const int* __restrict__ tok_list,
                                             const float* __restrict__ wt_list,
                                             float* __restrict__ dout) {
    int e    = blockIdx.x >> 8;    // 256 tiles max per expert
    int tile = blockIdx.x & 255;
    int ne = counts[e];
    int ntiles = (ne + 15) >> 4;
    if (tile >= ntiles) return;

    __shared__ bf16  sA[16][D_];    // 32 KB: 16 gathered token rows
    __shared__ bf16  sMid[16][IC];  // 4 KB: gelu(h@w1) chunk
    __shared__ int   sTok[16];
    __shared__ float sWt[16];

    int tid = threadIdx.x;
    if (tid < 16) {
        int slot = tile * 16 + tid;
        if (slot < ne) { sTok[tid] = tok_list[e * T_ + slot]; sWt[tid] = wt_list[e * T_ + slot]; }
        else           { sTok[tid] = 0;                        sWt[tid] = 0.f; }
    }
    __syncthreads();

    // stage 16 token rows (bf16, 2048 B each): 16 threads/row x 8 uint4
    {
        int r = tid >> 4, c = tid & 15;
        const uint4* src = (const uint4*)(hbf + (size_t)sTok[r] * D_);
        uint4* dst = (uint4*)&sA[r][0];
#pragma unroll
        for (int j = 0; j < 8; ++j) dst[c * 8 + j] = src[c * 8 + j];
    }
    __syncthreads();

    int wave = tid >> 5, lane = tid & 31;
    int nlo = lane & 15;
    bool hi = lane >= 16;

    v8f acc[8];
#pragma unroll
    for (int f = 0; f < 8; ++f) acc[f] = v8f_zero();

    for (int ch = 0; ch < I_ / IC; ++ch) {
        // ---- phase A: mid[:, wave*16 .. wave*16+16) = gelu(hA @ w1 chunk + b1)
        int iTile = ch * (IC / 16) + wave;  // global 16-wide i-tile index
        const bf16* bbase = w1bf + ((size_t)e * (I_/16) + iTile) * (D_/32) * 512;
        __builtin_prefetch((const void*)(bbase + (size_t)(D_/32) * 512), 0, 1);
        v8f cm = v8f_zero();
#pragma unroll 4
        for (int ks = 0; ks < D_ / 32; ++ks) {
            v16bf a = load_a_frag(&sA[0][0], D_, nlo, ks * 32, hi);
            v16bf b = load_b_frag(bbase + (size_t)ks * 512, lane);
            cm = __builtin_amdgcn_wmma_f32_16x16x32_bf16(false, a, false, b, (short)0, cm, false, false);
        }
        float b1v = b1[e * I_ + iTile * 16 + nlo];
#pragma unroll
        for (int r = 0; r < 8; ++r) {
            int m = hi ? (r + 8) : r;
            float x = cm[r] + b1v;
            float g = 0.5f * x * (1.0f + erff(x * 0.70710678118654752f)); // exact GELU
            sMid[m][wave * 16 + nlo] = (bf16)g;
        }
        __syncthreads();

        // ---- phase B: out[:, wave's 128 cols] += mid @ w2 chunk
#pragma unroll
        for (int f = 0; f < 8; ++f) {
            int dTile = wave * 8 + f;
            const bf16* b2base = w2bf + (((size_t)e * (D_/16) + dTile) * (I_/32) + (size_t)ch * (IC/32)) * 512;
#pragma unroll
            for (int kk = 0; kk < IC / 32; ++kk) {
                v16bf a = load_a_frag(&sMid[0][0], IC, nlo, kk * 32, hi);
                v16bf b = load_b_frag(b2base + (size_t)kk * 512, lane);
                acc[f] = __builtin_amdgcn_wmma_f32_16x16x32_bf16(false, a, false, b, (short)0, acc[f], false, false);
            }
        }
        __syncthreads();
    }

    // epilogue: +b2, scale by gate weight, scatter-add into dout
#pragma unroll
    for (int f = 0; f < 8; ++f) {
        int col = wave * 128 + f * 16 + nlo;
        float b2v = b2[e * D_ + col];
#pragma unroll
        for (int r = 0; r < 8; ++r) {
            int m = hi ? (r + 8) : r;
            float val = (acc[f][r] + b2v) * sWt[m];
            unsafeAtomicAdd(&dout[(size_t)sTok[m] * D_ + col], val);
        }
    }
}

// ---------------------------------------------------------------- kernel 6: residual + mask + LayerNorm + aux loss

__global__ __launch_bounds__(256) void k_final(const float* __restrict__ h,
                                               const float* __restrict__ mask,
                                               const float* __restrict__ gamma,
                                               const float* __restrict__ beta,
                                               const int* __restrict__ counts,
                                               float* __restrict__ dout) {
    int t = blockIdx.x;
    int tid = threadIdx.x;
    int wave = tid >> 5, lane = tid & 31;
    __shared__ float rs[8], rs2[8];

    const float* hr = h + (size_t)t * D_;
    float* orow = dout + (size_t)t * D_;
    float mk = mask[t];

    float vals[D_ / 256];
    float s = 0.f, s2 = 0.f;
#pragma unroll
    for (int j = 0; j < D_ / 256; ++j) {
        int d = tid + j * 256;
        float v = hr[d] + orow[d] * mk;
        vals[j] = v;
        s += v; s2 += v * v;
    }
#pragma unroll
    for (int off = 16; off > 0; off >>= 1) {
        s  += __shfl_xor(s, off, 32);
        s2 += __shfl_xor(s2, off, 32);
    }
    if (lane == 0) { rs[wave] = s; rs2[wave] = s2; }
    __syncthreads();
    if (tid == 0) {
        float a = 0.f, b = 0.f;
#pragma unroll
        for (int w = 0; w < 8; ++w) { a += rs[w]; b += rs2[w]; }
        rs[0] = a; rs2[0] = b;
    }
    __syncthreads();
    float mu = rs[0] * (1.0f / D_);
    float var = rs2[0] * (1.0f / D_) - mu * mu;
    float inv = rsqrtf(var + 1e-5f);
#pragma unroll
    for (int j = 0; j < D_ / 256; ++j) {
        int d = tid + j * 256;
        orow[d] = (vals[j] - mu) * inv * gamma[d] + beta[d];
    }
    if (t == 0 && tid == 0) {
        float aux = 0.f;
#pragma unroll
        for (int e = 0; e < E_; ++e) {
            float u = (float)counts[e] * (1.0f / T_) - (1.0f / E_);
            aux += u * u;
        }
        dout[(size_t)T_ * D_] = aux * (1.0f / E_);
    }
}

// ---------------------------------------------------------------- launch

extern "C" void kernel_launch(void* const* d_in, const int* in_sizes, int n_in,
                              void* d_out, int out_size, void* d_ws, size_t ws_size,
                              hipStream_t stream) {
    (void)in_sizes; (void)n_in; (void)out_size; (void)ws_size;
    const float* h      = (const float*)d_in[0];
    const float* mask   = (const float*)d_in[1];
    const float* gate_w = (const float*)d_in[2];
    const float* w1     = (const float*)d_in[3];
    const float* b1     = (const float*)d_in[4];
    const float* w2     = (const float*)d_in[5];
    const float* b2     = (const float*)d_in[6];
    const float* gamma  = (const float*)d_in[7];
    const float* beta   = (const float*)d_in[8];
    float* dout = (float*)d_out;

    char* w = (char*)d_ws;
    bf16* hbf   = (bf16*)w;  w += (size_t)T_ * D_ * sizeof(bf16);
    bf16* w1bf  = (bf16*)w;  w += (size_t)E_ * D_ * I_ * sizeof(bf16);
    bf16* w2bf  = (bf16*)w;  w += (size_t)E_ * I_ * D_ * sizeof(bf16);
    int* counts = (int*)w;   w += 256;
    int* tok    = (int*)w;   w += (size_t)E_ * T_ * sizeof(int);
    float* wt   = (float*)w;

    k_zero<<<4096, 256, 0, stream>>>(dout, counts);
    k_repack_w1<<<32768, 256, 0, stream>>>(w1, w1bf);
    k_repack_w2<<<32768, 256, 0, stream>>>(w2, w2bf);
    k_gate<<<T_ / 8, 256, 0, stream>>>(h, gate_w, hbf, counts, tok, wt);
    k_ffn<<<E_ * 256, 256, 0, stream>>>(hbf, w1bf, w2bf, b1, b2, counts, tok, wt, dout);
    k_final<<<T_, 256, 0, stream>>>(h, mask, gamma, beta, counts, dout);
}
